// LanguageEncoder_71485435675188
// MI455X (gfx1250) — matrix operand order repaired and checked by reading, not compiled, not run on hardware
//
#include <hip/hip_runtime.h>
#include <hip/hip_bf16.h>
#include <stdint.h>

#define T_STEPS 64
#define BATCH   256
#define DDIM    512
#define HDIM    1024
#define H3      (3 * HDIM)

typedef __attribute__((ext_vector_type(16))) __bf16 v16bf;
typedef __attribute__((ext_vector_type(8)))  __bf16 v8bf;
typedef __attribute__((ext_vector_type(8)))  float  v8f;

union ABu { v16bf v; v8bf h[2]; };

// A-operand tile (16x32 bf16): rowptr = lane's row + kk.
// elements 0..7 = K kb..kb+7, elements 8..15 = K kb+16..kb+23 (kb = half*8)
__device__ __forceinline__ v16bf load_a(const __bf16* rowptr, int kb) {
    ABu a;
    a.h[0] = *reinterpret_cast<const v8bf*>(rowptr + kb);
    a.h[1] = *reinterpret_cast<const v8bf*>(rowptr + kb + 16);
    return a.v;
}

// B-operand tile (32x16 bf16): p = lane's column + half*8; elements 0..7 = K p..p+7, 8..15 = K p+16..p+23.
__device__ __forceinline__ v16bf load_b(const __bf16* p) {
    ABu b;
    b.h[0] = *reinterpret_cast<const v8bf*>(p);
    b.h[1] = *reinterpret_cast<const v8bf*>(p + 16);
    return b.v;
}

__device__ __forceinline__ v8f wmma_bf16(v16bf a, v16bf b, v8f c) {
    return __builtin_amdgcn_wmma_f32_16x16x32_bf16(false, a, false, b,
                                                   (short)0, c, false, false);
}

// ---------------------------------------------------------------------------
// Panel GEMM: 8 waves, block macro-tile 128(M) x 64(N), wave tile 32x32.
// B panel (64 cols x 64 K bf16 = 8 KB) double-buffered in LDS, filled with
// global_load_async_to_lds_b128 (ASYNCcnt) and consumed via ds_load_b128.
// A rows are read directly from global (b128).
template<int KDIM>
__device__ __forceinline__ void gemm_panel(v8f (&c)[2][2],
        const __bf16* __restrict__ a0, const __bf16* __restrict__ a1,
        const __bf16* __restrict__ bpanel,   // wT + colbase*KDIM (64-col panel)
        __bf16* sB,                          // LDS: 2 x 64 x 64 bf16
        int kb, int half, int cl0, int cl1)  // local cols: wn*32+l, +16
{
    const int tid  = threadIdx.x;
    const int colp = tid >> 3;   // 0..31
    const int part = tid & 7;    // 0..7 (16-byte segment within a 64-K row)

    // Issue one 8KB panel stage: 512 b128 segments, 2 per thread.
    auto stage = [&](int kc, int buf) {
        const __bf16* g0 = bpanel + (size_t)colp * KDIM + kc + part * 8;
        const __bf16* g1 = g0 + (size_t)32 * KDIM;
        unsigned l0 = (unsigned)(uintptr_t)(sB) + (unsigned)(buf * 64 * 64 * 2)
                      + (unsigned)((colp * 64 + part * 8) * 2);
        unsigned l1 = l0 + 32u * 64u * 2u;
        asm volatile("global_load_async_to_lds_b128 %0, %1, off"
                     :: "v"(l0), "v"((unsigned long long)(uintptr_t)g0) : "memory");
        asm volatile("global_load_async_to_lds_b128 %0, %1, off"
                     :: "v"(l1), "v"((unsigned long long)(uintptr_t)g1) : "memory");
    };

    stage(0, 0);
    int buf = 0;
    for (int kc = 0; kc < KDIM; kc += 64) {
        if (kc + 64 < KDIM) {
            stage(kc + 64, buf ^ 1);
            asm volatile("s_wait_asynccnt 2" ::: "memory");  // stage(kc) done
        } else {
            asm volatile("s_wait_asynccnt 0" ::: "memory");
        }
        __syncthreads();  // panel visible to all waves
        const __bf16* lb = sB + buf * (64 * 64);
        #pragma unroll
        for (int k2 = 0; k2 < 64; k2 += 32) {
            v16bf A0 = load_a(a0 + kc + k2, kb);
            v16bf A1 = load_a(a1 + kc + k2, kb);
            v16bf B0 = load_b(lb + cl0 * 64 + k2 + half * 8);
            v16bf B1 = load_b(lb + cl1 * 64 + k2 + half * 8);
            c[0][0] = wmma_bf16(A0, B0, c[0][0]);
            c[0][1] = wmma_bf16(A0, B1, c[0][1]);
            c[1][0] = wmma_bf16(A1, B0, c[1][0]);
            c[1][1] = wmma_bf16(A1, B1, c[1][1]);
        }
        __syncthreads();  // protect buffer before next overwrite
        buf ^= 1;
    }
}

// ---------------------------------------------------------------------------
__global__ __launch_bounds__(256) void kInit(float* hfp, __bf16* hbf, float* out) {
    int i = blockIdx.x * blockDim.x + threadIdx.x;
    if (i < BATCH * HDIM) {
        hfp[i] = 0.0f;
        hbf[i] = (__bf16)0.0f;
        out[i] = 0.0f;
    }
}

__global__ __launch_bounds__(256) void kPack(const float* __restrict__ w_i,
                                             const float* __restrict__ w_h,
                                             __bf16* __restrict__ wiT,
                                             __bf16* __restrict__ whT) {
    int n = blockIdx.x;  // 0 .. 3071
    for (int k = threadIdx.x; k < DDIM; k += blockDim.x)
        wiT[(size_t)n * DDIM + k] = (__bf16)w_i[(size_t)k * H3 + n];
    for (int k = threadIdx.x; k < HDIM; k += blockDim.x)
        whT[(size_t)n * HDIM + k] = (__bf16)w_h[(size_t)k * H3 + n];
}

__global__ __launch_bounds__(256) void kEmbed(const int* __restrict__ x,
                                              const float* __restrict__ embed,
                                              __bf16* __restrict__ words) {
    int row = blockIdx.x;  // t*B + b
    int tok = x[row];
    float scale = (tok > 0) ? 1.0f : 0.0f;
    const float* e = embed + (size_t)tok * DDIM;
    for (int d = threadIdx.x; d < DDIM; d += blockDim.x)
        words[(size_t)row * DDIM + d] = (__bf16)(e[d] * scale);
}

// ---------------------------------------------------------------------------
// Step Z: zr = sigmoid(words_t @ w_i[:, :2H] + h @ w_h[:, :2H] + b)
// Writes z (f32) and rh = r*h (bf16). Grid: 64 blocks x 256 threads.
__global__ __launch_bounds__(256) void kStepZ(const __bf16* __restrict__ words,
                                              const __bf16* __restrict__ wiT,
                                              const __bf16* __restrict__ whT,
                                              const __bf16* __restrict__ hbf,
                                              const float*  __restrict__ hfp,
                                              const float*  __restrict__ bias,
                                              float*  __restrict__ zbuf,
                                              __bf16* __restrict__ rhbuf,
                                              int t) {
    __shared__ __bf16 sB[2][64 * 64];
    const int tid  = threadIdx.x;
    const int lane = tid & 31, wave = tid >> 5;
    const int wm = wave & 3, wn = wave >> 2;
    const int half = lane >> 4, l = lane & 15;
    const int bm = blockIdx.x & 1;      // 2 M macro-blocks (256/128)
    const int bn = blockIdx.x >> 1;     // 32 N macro-blocks (2048/64)
    const int mbase = bm * 128 + wm * 32;
    const int nbg   = bn * 64;          // global column base of block panel
    const int cl0 = wn * 32 + l, cl1 = cl0 + 16;
    const int kb = half * 8;
    const int arow0 = mbase + l, arow1 = mbase + 16 + l;

    v8f c[2][2] = {};

    // Phase 1: words_t @ w_i (K = 512)
    gemm_panel<DDIM>(c,
        words + (size_t)(t * BATCH + arow0) * DDIM,
        words + (size_t)(t * BATCH + arow1) * DDIM,
        wiT + (size_t)nbg * DDIM, &sB[0][0], kb, half, cl0, cl1);
    // Phase 2: h @ w_h (K = 1024)
    gemm_panel<HDIM>(c,
        hbf + (size_t)arow0 * HDIM,
        hbf + (size_t)arow1 * HDIM,
        whT + (size_t)nbg * HDIM, &sB[0][0], kb, half, cl0, cl1);

    #pragma unroll
    for (int i = 0; i < 2; ++i) {
        #pragma unroll
        for (int j = 0; j < 2; ++j) {
            int colg = nbg + cl0 + j * 16;  // zr column in [0, 2048)
            float bb = bias[colg];
            #pragma unroll
            for (int r = 0; r < 8; ++r) {
                int row = mbase + i * 16 + half * 8 + r;
                float s = 1.0f / (1.0f + __expf(-(c[i][j][r] + bb)));
                if (colg < HDIM) {
                    zbuf[(size_t)row * HDIM + colg] = s;
                } else {
                    int jj = colg - HDIM;
                    float hv = hfp[(size_t)row * HDIM + jj];
                    rhbuf[(size_t)row * HDIM + jj] = (__bf16)(s * hv);
                }
            }
        }
    }
}

// ---------------------------------------------------------------------------
// Step A: a = tanh(words_t @ w_i[:,2H:] + rh @ w_h[:,2H:] + b_a)
// h = (1-z)h + z*a ; out += h. Grid: 32 blocks x 256 threads.
__global__ __launch_bounds__(256) void kStepA(const __bf16* __restrict__ words,
                                              const __bf16* __restrict__ wiT,
                                              const __bf16* __restrict__ whT,
                                              const __bf16* __restrict__ rhbuf,
                                              const float*  __restrict__ zbuf,
                                              float*  __restrict__ hfp,
                                              __bf16* __restrict__ hbf,
                                              const float* __restrict__ bias,
                                              float* __restrict__ out,
                                              int t) {
    __shared__ __bf16 sB[2][64 * 64];
    const int tid  = threadIdx.x;
    const int lane = tid & 31, wave = tid >> 5;
    const int wm = wave & 3, wn = wave >> 2;
    const int half = lane >> 4, l = lane & 15;
    const int bm = blockIdx.x & 1;      // 2 M macro-blocks
    const int bn = blockIdx.x >> 1;     // 16 N macro-blocks (1024/64)
    const int mbase = bm * 128 + wm * 32;
    const int nbg   = bn * 64;          // a-column base in [0, 1024)
    const int cl0 = wn * 32 + l, cl1 = cl0 + 16;
    const int kb = half * 8;
    const int arow0 = mbase + l, arow1 = mbase + 16 + l;
    const int wbase = 2 * HDIM + nbg;   // weight/bias column base

    v8f c[2][2] = {};

    // Phase 1: words_t @ w_i[:, 2H:] (K = 512)
    gemm_panel<DDIM>(c,
        words + (size_t)(t * BATCH + arow0) * DDIM,
        words + (size_t)(t * BATCH + arow1) * DDIM,
        wiT + (size_t)wbase * DDIM, &sB[0][0], kb, half, cl0, cl1);
    // Phase 2: (r*h) @ w_h[:, 2H:] (K = 1024)
    gemm_panel<HDIM>(c,
        rhbuf + (size_t)arow0 * HDIM,
        rhbuf + (size_t)arow1 * HDIM,
        whT + (size_t)wbase * HDIM, &sB[0][0], kb, half, cl0, cl1);

    #pragma unroll
    for (int i = 0; i < 2; ++i) {
        #pragma unroll
        for (int j = 0; j < 2; ++j) {
            int colg = nbg + cl0 + j * 16;  // a-column in [0, 1024)
            float bb = bias[2 * HDIM + colg];
            #pragma unroll
            for (int r = 0; r < 8; ++r) {
                int row = mbase + i * 16 + half * 8 + r;
                size_t idx = (size_t)row * HDIM + colg;
                float av = tanhf(c[i][j][r] + bb);
                float zv = zbuf[idx];
                float hv = hfp[idx];
                float hn = (1.0f - zv) * hv + zv * av;
                hfp[idx] = hn;
                hbf[idx] = (__bf16)hn;
                out[idx] += hn;
            }
        }
    }
}

// ---------------------------------------------------------------------------
extern "C" void kernel_launch(void* const* d_in, const int* in_sizes, int n_in,
                              void* d_out, int out_size, void* d_ws, size_t ws_size,
                              hipStream_t stream) {
    const int*   x     = (const int*)d_in[0];     // (T, B) int32
    const float* embed = (const float*)d_in[1];   // (VOCAB, D)
    const float* w_i   = (const float*)d_in[2];   // (D, 3H)
    const float* w_h   = (const float*)d_in[3];   // (H, 3H)
    const float* bias  = (const float*)d_in[4];   // (3H,)
    float* out = (float*)d_out;                   // (B, H) f32

    char* p = (char*)d_ws;
    __bf16* words = (__bf16*)p; p += (size_t)T_STEPS * BATCH * DDIM * 2; // 16 MiB
    __bf16* wiT   = (__bf16*)p; p += (size_t)H3 * DDIM * 2;             // 3 MiB
    __bf16* whT   = (__bf16*)p; p += (size_t)H3 * HDIM * 2;             // 6 MiB
    float*  hfp   = (float*)p;  p += (size_t)BATCH * HDIM * 4;
    __bf16* hbf   = (__bf16*)p; p += (size_t)BATCH * HDIM * 2;
    float*  zbuf  = (float*)p;  p += (size_t)BATCH * HDIM * 4;
    __bf16* rhbuf = (__bf16*)p; p += (size_t)BATCH * HDIM * 2;

    hipLaunchKernelGGL(kInit, dim3((BATCH * HDIM + 255) / 256), dim3(256), 0, stream,
                       hfp, hbf, out);
    hipLaunchKernelGGL(kPack, dim3(H3), dim3(256), 0, stream, w_i, w_h, wiT, whT);
    hipLaunchKernelGGL(kEmbed, dim3(T_STEPS * BATCH), dim3(256), 0, stream,
                       x, embed, words);

    for (int t = 0; t < T_STEPS; ++t) {
        hipLaunchKernelGGL(kStepZ, dim3(64), dim3(256), 0, stream,
                           words, wiT, whT, hbf, hfp, bias, zbuf, rhbuf, t);
        hipLaunchKernelGGL(kStepA, dim3(32), dim3(256), 0, stream,
                           words, wiT, whT, rhbuf, zbuf, hfp, hbf, bias, out, t);
    }
}
